// FlexibleCrystalGraphVAE_56848187129900
// MI455X (gfx1250) — matrix operand order repaired and checked by reading, not compiled, not run on hardware
//
#include <hip/hip_runtime.h>

typedef __attribute__((ext_vector_type(16))) _Float16 v16h;
typedef __attribute__((ext_vector_type(8)))  _Float16 v8h;
typedef __attribute__((ext_vector_type(8)))  float    v8f;

#define NTOT  40000
#define MNBR  12
#define AFL   64
#define NBR   41
#define HID   128
#define LATD  64
#define BCRYS 800
#define ORIG  92
#define NA    20
#define NPAIR 380   // NA*(NA-1)

__device__ __forceinline__ float sigmoidf_(float x){ return 1.0f/(1.0f+__expf(-x)); }
__device__ __forceinline__ float softplusf_(float x){ return (x > 20.f) ? x : log1pf(__expf(x)); }

__device__ __forceinline__ v8f wmma_f16(v16h a, v16h b, v8f c){
  // v_wmma_f32_16x16x32_f16: (neg_a, A, neg_b, B, c_mod, C, reuse_a, reuse_b)
  return __builtin_amdgcn_wmma_f32_16x16x32_f16(false, a, false, b, (short)0, c, false, false);
}

// A fragment (16x32 f16) from LDS row-major tile; per CDNA5 ISA 16-bit A layout:
// lane&15 = row M, koff = (lane&16)?8:0; elems 0..7 -> K=koff.. , elems 8..15 -> K=16+koff..
__device__ __forceinline__ v16h load_a_frag(const _Float16* p, int koff){
  v8h lo = *(const v8h*)(p + koff);
  v8h hi = *(const v8h*)(p + 16 + koff);
  return __builtin_shufflevector(lo, hi, 0,1,2,3,4,5,6,7,8,9,10,11,12,13,14,15);
}

// ---------------- weight pre-pack into B fragment layout ----------------
// B (32x16 KxN) layout: lane&15 = col N, K = (e&8?16:0)+(lane&16?8:0)+(e&7)  (e = 0..15 within v16h)
__global__ void pack_b_kernel(const float* __restrict__ W, _Float16* __restrict__ dst,
                              int K, int Ncol, int coltiles, int chunks){
  int idx = blockIdx.x*256 + threadIdx.x;
  int total = coltiles*chunks*32*16;
  if (idx >= total) return;
  int e    = idx & 15;
  int lane = (idx >> 4) & 31;
  int t    = idx >> 9;            // ct*chunks + chunk
  int chunk = t % chunks;
  int ct    = t / chunks;
  int n = ct*16 + (lane & 15);
  int k = chunk*32 + ((e & 8) ? 16 : 0) + ((lane & 16) ? 8 : 0) + (e & 7);
  float v = (k < K && n < Ncol) ? W[k*Ncol + n] : 0.0f;
  dst[idx] = (_Float16)v;
}

// ---------------- embedding: x = atom_fea @ W_embed + b ----------------
__global__ void embed_kernel(const float* __restrict__ atom_fea, const float* __restrict__ W,
                             const float* __restrict__ bias, float* __restrict__ x,
                             _Float16* __restrict__ xh){
  __shared__ float arow[ORIG];
  int i = blockIdx.x, c = threadIdx.x;   // 64 threads
  for (int k = c; k < ORIG; k += 64) arow[k] = atom_fea[i*ORIG + k];
  __syncthreads();
  float s = bias[c];
  #pragma unroll 4
  for (int k = 0; k < ORIG; ++k) s = fmaf(arow[k], W[k*AFL + c], s);
  x[i*AFL + c] = s;
  xh[i*AFL + c] = (_Float16)s;
}

// ---------------- conv GEMM (WMMA), PASS 1: BN stats, PASS 2: apply+reduce ----------------
template<int PASS>
__global__ void conv_gemm_kernel(const _Float16* __restrict__ xh, const float* __restrict__ nbr_fea,
                                 const int* __restrict__ nbr_idx, const _Float16* __restrict__ pB,
                                 const float* __restrict__ bc,
                                 const float* __restrict__ mean1, const float* __restrict__ rstd1,
                                 const float* __restrict__ g1, const float* __restrict__ b1,
                                 float* __restrict__ part, float* __restrict__ summed){
  __shared__ __align__(16) _Float16 Al[16*192];   // A tile, K padded 169->192
  __shared__ float xch[16*128];                   // PASS2: normalized gated tile
  __shared__ int atomi[16], nbrs[16];
  int tid = threadIdx.x, lane = tid & 31, wave = tid >> 5;
  int base = blockIdx.x * 16;                     // rows over N*M
  if (tid < 16){
    int grow = base + tid;
    atomi[tid] = grow / MNBR;
    nbrs[tid]  = nbr_idx[grow];
  }
  __syncthreads();
  // A-tile staging in 16-byte chunks: row = 24 chunks of 8 halves
  for (int q = tid; q < 16*24; q += 128){
    int r = q / 24, ch = q - r*24;
    v8h v;
    if (ch < 8){
      v = *(const v8h*)(xh + atomi[r]*AFL + ch*8);
    } else if (ch < 16){
      v = *(const v8h*)(xh + nbrs[r]*AFL + (ch-8)*8);
    } else {
      int cbase = 128 + (ch-16)*8;
      int grow = base + r;
      #pragma unroll
      for (int t = 0; t < 8; ++t){
        int c = cbase + t;
        v[t] = (c < 169) ? (_Float16)nbr_fea[grow*NBR + (c-128)] : (_Float16)0.0f;
      }
    }
    *(v8h*)(Al + r*192 + ch*8) = v;
  }
  __syncthreads();
  int row  = lane & 15;
  int koff = (lane & 16) ? 8 : 0;
  int rbase = (lane & 16) ? 8 : 0;
  v8f acc0 = {}, acc1 = {};
  #pragma unroll
  for (int ch = 0; ch < 6; ++ch){
    v16h a  = load_a_frag(&Al[row*192 + ch*32], koff);
    v16h b0 = *(const v16h*)(pB + (((wave*2+0)*6 + ch)*32 + lane)*16);
    v16h b1v= *(const v16h*)(pB + (((wave*2+1)*6 + ch)*32 + lane)*16);
    acc0 = wmma_f16(a, b0, acc0);
    acc1 = wmma_f16(a, b1v, acc1);
  }
  #pragma unroll
  for (int tt = 0; tt < 2; ++tt){
    v8f acc = tt ? acc1 : acc0;
    int col = (wave*2 + tt)*16 + (lane & 15);
    if (PASS == 1){
      float s = 0.f, s2 = 0.f;
      #pragma unroll
      for (int v = 0; v < 8; ++v){ float val = acc[v] + bc[col]; s += val; s2 += val*val; }
      s  += __shfl_xor(s, 16);
      s2 += __shfl_xor(s2, 16);
      if (lane < 16){
        float* pp = part + (blockIdx.x & 63) * 256;
        atomicAdd(pp + col, s);
        atomicAdd(pp + 128 + col, s2);
      }
    } else {
      float m = mean1[col], r = rstd1[col], gg = g1[col], bb = b1[col];
      #pragma unroll
      for (int v = 0; v < 8; ++v)
        xch[(rbase + v)*128 + col] = (acc[v] + bc[col] - m) * r * gg + bb;
    }
  }
  if (PASS == 2){
    __syncthreads();
    // run-length accumulate over rows sharing the same atom before global atomics
    int c = tid & 63;
    int rstart = (tid >> 6) * 8;     // rows 0..7 or 8..15
    float accum = 0.0f;
    int cur = atomi[rstart];
    #pragma unroll
    for (int r = rstart; r < rstart + 8; ++r){
      int a = atomi[r];
      if (a != cur){
        atomicAdd(summed + cur*AFL + c, accum);
        accum = 0.0f; cur = a;
      }
      accum += sigmoidf_(xch[r*128 + c]) * softplusf_(xch[r*128 + 64 + c]);
    }
    atomicAdd(summed + cur*AFL + c, accum);
  }
}

// ---------------- column stats over [rows x C] matrix ----------------
__global__ void colstats_kernel(const float* __restrict__ X, int rows, int C,
                                float* __restrict__ part){
  int tid = threadIdx.x;           // 256 threads, C must divide 256 (64 or 128)
  int col = tid % C;
  int rg  = tid / C;
  int rgs = 256 / C;
  float s = 0.f, s2 = 0.f;
  for (int r = blockIdx.x*rgs + rg; r < rows; r += gridDim.x*rgs){
    float v = X[r*C + col]; s += v; s2 += v*v;
  }
  __shared__ float ls[256], ls2[256];
  ls[tid] = s; ls2[tid] = s2;
  __syncthreads();
  if (rg == 0){
    for (int g = 1; g < rgs; ++g){ s += ls[g*C + col]; s2 += ls2[g*C + col]; }
    float* pp = part + (blockIdx.x & 63)*256;
    atomicAdd(pp + col, s);
    atomicAdd(pp + 128 + col, s2);
  }
}

__global__ void finalize_stats_kernel(const float* __restrict__ part, float* __restrict__ mean,
                                      float* __restrict__ rstd, int C, float rows){
  int c = blockIdx.x*64 + threadIdx.x;
  if (c >= C) return;
  float s = 0.f, s2 = 0.f;
  for (int b = 0; b < 64; ++b){ s += part[b*256 + c]; s2 += part[b*256 + 128 + c]; }
  float m = s / rows;
  float var = s2 / rows - m*m;
  mean[c] = m;
  rstd[c] = rsqrtf(var + 1e-5f);
}

// ---------------- conv residual update: x = softplus(x + bn2(summed)) ----------------
__global__ void conv_update_kernel(float* __restrict__ x, _Float16* __restrict__ xh,
                                   const float* __restrict__ summed, const float* __restrict__ mean,
                                   const float* __restrict__ rstd, const float* __restrict__ g2,
                                   const float* __restrict__ b2){
  int idx = blockIdx.x*256 + threadIdx.x;
  if (idx >= NTOT*AFL) return;
  int c = idx & 63;
  float v = x[idx] + (summed[idx] - mean[c]) * rstd[c] * g2[c] + b2[c];
  float nx = softplusf_(v);
  x[idx] = nx;
  xh[idx] = (_Float16)nx;
}

// ---------------- segment-mean pooling ----------------
__global__ void pool_accum_kernel(const float* __restrict__ x, const int* __restrict__ cidx,
                                  float* __restrict__ csum, float* __restrict__ cnt){
  int idx = blockIdx.x*256 + threadIdx.x;
  if (idx >= NTOT*AFL) return;
  int i = idx >> 6, c = idx & 63;
  int b = cidx[i];
  atomicAdd(csum + b*AFL + c, x[idx]);
  if (c == 0) atomicAdd(cnt + b, 1.0f);
}

__global__ void pool_div_kernel(const float* __restrict__ csum, const float* __restrict__ cnt,
                                float* __restrict__ crys){
  int idx = blockIdx.x*256 + threadIdx.x;
  if (idx >= BCRYS*AFL) return;
  crys[idx] = csum[idx] / fmaxf(cnt[idx >> 6], 1.0f);
}

// ---------------- small row GEMMs ----------------
__global__ void gemm64to128_kernel(const float* __restrict__ in, const float* __restrict__ W,
                                   const float* __restrict__ bias, float* __restrict__ out){
  __shared__ float rowv[64];
  int b = blockIdx.x, c = threadIdx.x;   // 128 threads
  if (c < 64) rowv[c] = in[b*64 + c];
  __syncthreads();
  float s = bias[c];
  #pragma unroll 4
  for (int k = 0; k < 64; ++k) s = fmaf(rowv[k], W[k*128 + c], s);
  out[b*128 + c] = s;
}

__global__ void bn_relu_kernel(const float* __restrict__ pre, float* __restrict__ out,
                               const float* __restrict__ mean, const float* __restrict__ rstd,
                               const float* __restrict__ g, const float* __restrict__ beta,
                               int n, int C){
  int idx = blockIdx.x*256 + threadIdx.x;
  if (idx >= n) return;
  int c = idx % C;
  out[idx] = fmaxf((pre[idx] - mean[c]) * rstd[c] * g[c] + beta[c], 0.0f);
}

__global__ void heads_kernel(const float* __restrict__ comb, const float* __restrict__ Wmu,
                             const float* __restrict__ bmu, const float* __restrict__ Wlv,
                             const float* __restrict__ blv, const float* __restrict__ eps,
                             float* __restrict__ out_mu, float* __restrict__ out_lv,
                             float* __restrict__ z){
  __shared__ float rowv[128];
  int b = blockIdx.x, l = threadIdx.x;   // 64 threads
  rowv[l] = comb[b*128 + l];
  rowv[l + 64] = comb[b*128 + l + 64];
  __syncthreads();
  float smu = bmu[l], slv = blv[l];
  #pragma unroll 4
  for (int k = 0; k < 128; ++k){
    float cv = rowv[k];
    smu = fmaf(cv, Wmu[k*64 + l], smu);
    slv = fmaf(cv, Wlv[k*64 + l], slv);
  }
  out_mu[b*64 + l] = smu;
  out_lv[b*64 + l] = slv;
  z[b*64 + l] = smu + eps[b*64 + l] * __expf(0.5f * slv);
}

__global__ void af1_kernel(const float* __restrict__ sfeat, const float* __restrict__ Wa1,
                           const float* __restrict__ ba1, float* __restrict__ af1){
  __shared__ float rowv[128];
  int r = blockIdx.x;                // 0..B*NA
  int b = r / NA, a = r - b*NA;
  int c = threadIdx.x;               // 128 threads
  rowv[c] = sfeat[b*128 + c];
  __syncthreads();
  float s = ba1[c] + ((float)a / (float)NA) * Wa1[128*128 + c];   // pos channel (k=128)
  #pragma unroll 4
  for (int k = 0; k < 128; ++k) s = fmaf(rowv[k], Wa1[k*128 + c], s);
  af1[r*128 + c] = fmaxf(s, 0.0f);
}

__global__ void af2_kernel(const float* __restrict__ af1, const float* __restrict__ Wa2,
                           const float* __restrict__ ba2, float* __restrict__ out_af,
                           _Float16* __restrict__ afh){
  __shared__ float rowv[128];
  int r = blockIdx.x, c = threadIdx.x;   // 64 threads
  rowv[c] = af1[r*128 + c];
  rowv[c + 64] = af1[r*128 + c + 64];
  __syncthreads();
  float s = ba2[c];
  #pragma unroll 4
  for (int k = 0; k < 128; ++k) s = fmaf(rowv[k], Wa2[k*64 + c], s);
  float v = fmaxf(s, 0.0f);
  out_af[r*64 + c] = v;
  afh[r*64 + c] = (_Float16)v;
}

__global__ void t1_kernel(const float* __restrict__ out_af, const float* __restrict__ Wt1,
                          const float* __restrict__ bt1, float* __restrict__ t1){
  __shared__ float rowv[64];
  int r = blockIdx.x, c = threadIdx.x;   // 64 threads
  rowv[c] = out_af[r*64 + c];
  __syncthreads();
  float s = bt1[c];
  #pragma unroll 4
  for (int k = 0; k < 64; ++k) s = fmaf(rowv[k], Wt1[k*64 + c], s);
  t1[r*64 + c] = fmaxf(s, 0.0f);
}

__global__ void at_softmax_kernel(const float* __restrict__ t1, const float* __restrict__ Wt2,
                                  const float* __restrict__ bt2, float* __restrict__ out_at){
  __shared__ float rowv[64];
  __shared__ float vals[128];
  __shared__ float red[128];
  int r = blockIdx.x, c = threadIdx.x;   // 128 threads
  if (c < 64) rowv[c] = t1[r*64 + c];
  __syncthreads();
  float val = -3.0e38f;
  if (c < ORIG){
    float s = bt2[c];
    #pragma unroll 4
    for (int k = 0; k < 64; ++k) s = fmaf(rowv[k], Wt2[k*ORIG + c], s);
    val = s;
    vals[c] = s;
  }
  red[c] = val;
  __syncthreads();
  for (int st = 64; st > 0; st >>= 1){ if (c < st) red[c] = fmaxf(red[c], red[c + st]); __syncthreads(); }
  float mx = red[0];
  __syncthreads();
  float e = (c < ORIG) ? __expf(vals[c] - mx) : 0.0f;
  red[c] = e;
  __syncthreads();
  for (int st = 64; st > 0; st >>= 1){ if (c < st) red[c] += red[c + st]; __syncthreads(); }
  float sum = red[0];
  if (c < ORIG) out_at[r*ORIG + c] = e / sum;
}

// ---------------- fused bond decoder (WMMA x2) ----------------
__global__ void bonds_kernel(const _Float16* __restrict__ afh, const _Float16* __restrict__ pb1,
                             const float* __restrict__ bb1, const _Float16* __restrict__ pb2,
                             const float* __restrict__ bb2, float* __restrict__ out_bonds){
  __shared__ __align__(16) _Float16 Al[16*128];
  __shared__ __align__(16) _Float16 Hl[16*64];
  __shared__ int rowb[16], rowj[16], rowk[16];
  int tid = threadIdx.x, lane = tid & 31, wave = tid >> 5;
  int base = blockIdx.x * 16;           // rows over B*NPAIR = 304000
  if (tid < 16){
    int grow = base + tid;
    int b = grow / NPAIR;
    int p = grow - b*NPAIR;
    int j = p / (NA - 1);
    int kr = p - j*(NA - 1);
    rowb[tid] = b; rowj[tid] = j; rowk[tid] = kr + (kr >= j ? 1 : 0);
  }
  __syncthreads();
  // stage pair rows in 16-byte chunks: row = 16 chunks of 8 halves
  for (int q = tid; q < 16*16; q += 128){
    int r = q >> 4, ch = q & 15;
    v8h v;
    if (ch < 8) v = *(const v8h*)(afh + (rowb[r]*NA + rowj[r])*AFL + ch*8);
    else        v = *(const v8h*)(afh + (rowb[r]*NA + rowk[r])*AFL + (ch-8)*8);
    *(v8h*)(Al + r*128 + ch*8) = v;
  }
  __syncthreads();
  int row = lane & 15, koff = (lane & 16) ? 8 : 0, rbase = (lane & 16) ? 8 : 0;
  v8f acc = {};
  #pragma unroll
  for (int ch = 0; ch < 4; ++ch){
    v16h a  = load_a_frag(&Al[row*128 + ch*32], koff);
    v16h bf = *(const v16h*)(pb1 + ((wave*4 + ch)*32 + lane)*16);
    acc = wmma_f16(a, bf, acc);
  }
  int col = wave*16 + (lane & 15);
  #pragma unroll
  for (int v = 0; v < 8; ++v)
    Hl[(rbase + v)*64 + col] = (_Float16)fmaxf(acc[v] + bb1[col], 0.0f);
  __syncthreads();
  if (wave < 3){
    v8f acc2 = {};
    #pragma unroll
    for (int ch = 0; ch < 2; ++ch){
      v16h a  = load_a_frag(&Hl[row*64 + ch*32], koff);
      v16h bf = *(const v16h*)(pb2 + ((wave*2 + ch)*32 + lane)*16);
      acc2 = wmma_f16(a, bf, acc2);
    }
    int col2 = wave*16 + (lane & 15);
    if (col2 < NBR){
      #pragma unroll
      for (int v = 0; v < 8; ++v)
        out_bonds[(base + rbase + v)*NBR + col2] = sigmoidf_(acc2[v] + bb2[col2]);
    }
  }
}

extern "C" void kernel_launch(void* const* d_in, const int* in_sizes, int n_in,
                              void* d_out, int out_size, void* d_ws, size_t ws_size,
                              hipStream_t stream) {
  (void)in_sizes; (void)n_in; (void)out_size; (void)ws_size;
  const float* atom_fea = (const float*)d_in[0];
  const float* nbr_fea  = (const float*)d_in[1];
  const int*   nbr_idx  = (const int*)d_in[2];
  const int*   cidx     = (const int*)d_in[3];
  const float* eps      = (const float*)d_in[4];
  // d_in[5] = num_atoms (static NA=20)
  const float* W_embed = (const float*)d_in[6];
  const float* b_embed = (const float*)d_in[7];
  const float* Wc  = (const float*)d_in[8];
  const float* bc  = (const float*)d_in[9];
  const float* g1  = (const float*)d_in[10];
  const float* b1  = (const float*)d_in[11];
  const float* g2  = (const float*)d_in[12];
  const float* b2  = (const float*)d_in[13];
  const float* Wfc = (const float*)d_in[14];
  const float* bfc = (const float*)d_in[15];
  const float* gfc = (const float*)d_in[16];
  const float* betafc = (const float*)d_in[17];
  const float* Wmu = (const float*)d_in[18];
  const float* bmu = (const float*)d_in[19];
  const float* Wlv = (const float*)d_in[20];
  const float* blv = (const float*)d_in[21];
  const float* Wl  = (const float*)d_in[22];
  const float* bl  = (const float*)d_in[23];
  const float* gl  = (const float*)d_in[24];
  const float* betal = (const float*)d_in[25];
  const float* Wa1 = (const float*)d_in[26];
  const float* ba1 = (const float*)d_in[27];
  const float* Wa2 = (const float*)d_in[28];
  const float* ba2 = (const float*)d_in[29];
  const float* Wt1 = (const float*)d_in[30];
  const float* bt1 = (const float*)d_in[31];
  const float* Wt2 = (const float*)d_in[32];
  const float* bt2 = (const float*)d_in[33];
  const float* Wb1 = (const float*)d_in[34];
  const float* bb1 = (const float*)d_in[35];
  const float* Wb2 = (const float*)d_in[36];
  const float* bb2 = (const float*)d_in[37];

  float* out    = (float*)d_out;
  float* out_mu = out;
  float* out_lv = out + 51200;
  float* out_af = out + 102400;
  float* out_at = out + 1126400;
  float* out_bd = out + 2598400;

  char* ws = (char*)d_ws;
  size_t off = 0;
  auto alloc = [&](size_t bytes)->char* {
    char* p = ws + off;
    off += (bytes + 255) & ~(size_t)255;
    return p;
  };
  _Float16* xh     = (_Float16*)alloc((size_t)NTOT*AFL*2);
  float*    x      = (float*)alloc((size_t)NTOT*AFL*4);
  float*    summed = (float*)alloc((size_t)NTOT*AFL*4);
  float*    part   = (float*)alloc(64*256*4);
  float*    mean1  = (float*)alloc(128*4);
  float*    rstd1  = (float*)alloc(128*4);
  float*    mean2  = (float*)alloc(64*4);
  float*    rstd2  = (float*)alloc(64*4);
  float*    csum   = (float*)alloc(BCRYS*AFL*4);
  float*    cnt    = (float*)alloc(BCRYS*4);
  float*    crys   = (float*)alloc(BCRYS*AFL*4);
  float*    pre    = (float*)alloc(BCRYS*HID*4);
  float*    comb   = (float*)alloc(BCRYS*HID*4);
  float*    z      = (float*)alloc(BCRYS*LATD*4);
  float*    sfeat  = (float*)alloc(BCRYS*HID*4);
  float*    meanh  = (float*)alloc(128*4);
  float*    rstdh  = (float*)alloc(128*4);
  float*    af1    = (float*)alloc((size_t)BCRYS*NA*HID*4);
  _Float16* afh    = (_Float16*)alloc((size_t)BCRYS*NA*AFL*2);
  float*    t1     = (float*)alloc((size_t)BCRYS*NA*AFL*4);
  _Float16* pconv  = (_Float16*)alloc((size_t)3*8*6*32*16*2);
  _Float16* pb1    = (_Float16*)alloc((size_t)4*4*32*16*2);
  _Float16* pb2    = (_Float16*)alloc((size_t)3*2*32*16*2);

  // pack weights into WMMA B fragment layout
  for (int l = 0; l < 3; ++l)
    pack_b_kernel<<<(8*6*32*16 + 255)/256, 256, 0, stream>>>(
        Wc + l*169*128, pconv + l*8*6*32*16, 169, 128, 8, 6);
  pack_b_kernel<<<(4*4*32*16 + 255)/256, 256, 0, stream>>>(Wb1, pb1, 128, 64, 4, 4);
  pack_b_kernel<<<(3*2*32*16 + 255)/256, 256, 0, stream>>>(Wb2, pb2, 64, 41, 3, 2);

  embed_kernel<<<NTOT, 64, 0, stream>>>(atom_fea, W_embed, b_embed, x, xh);

  const int convBlocks = NTOT*MNBR/16;   // 30000
  for (int l = 0; l < 3; ++l){
    const _Float16* pB  = pconv + l*8*6*32*16;
    const float*    bcl = bc + l*128;
    hipMemsetAsync(part, 0, 64*256*4, stream);
    conv_gemm_kernel<1><<<convBlocks, 128, 0, stream>>>(
        xh, nbr_fea, nbr_idx, pB, bcl, nullptr, nullptr, nullptr, nullptr, part, nullptr);
    finalize_stats_kernel<<<2, 64, 0, stream>>>(part, mean1, rstd1, 128, (float)(NTOT*MNBR));
    hipMemsetAsync(summed, 0, (size_t)NTOT*AFL*4, stream);
    conv_gemm_kernel<2><<<convBlocks, 128, 0, stream>>>(
        xh, nbr_fea, nbr_idx, pB, bcl, mean1, rstd1, g1 + l*128, b1 + l*128, nullptr, summed);
    hipMemsetAsync(part, 0, 64*256*4, stream);
    colstats_kernel<<<256, 256, 0, stream>>>(summed, NTOT, 64, part);
    finalize_stats_kernel<<<1, 64, 0, stream>>>(part, mean2, rstd2, 64, (float)NTOT);
    conv_update_kernel<<<(NTOT*AFL + 255)/256, 256, 0, stream>>>(
        x, xh, summed, mean2, rstd2, g2 + l*64, b2 + l*64);
  }

  // pooling
  hipMemsetAsync(csum, 0, BCRYS*AFL*4, stream);
  hipMemsetAsync(cnt, 0, BCRYS*4, stream);
  pool_accum_kernel<<<(NTOT*AFL + 255)/256, 256, 0, stream>>>(x, cidx, csum, cnt);
  pool_div_kernel<<<(BCRYS*AFL + 255)/256, 256, 0, stream>>>(csum, cnt, crys);

  // combiner + BN + relu
  gemm64to128_kernel<<<BCRYS, 128, 0, stream>>>(crys, Wfc, bfc, pre);
  hipMemsetAsync(part, 0, 64*256*4, stream);
  colstats_kernel<<<256, 256, 0, stream>>>(pre, BCRYS, 128, part);
  finalize_stats_kernel<<<2, 64, 0, stream>>>(part, meanh, rstdh, 128, (float)BCRYS);
  bn_relu_kernel<<<(BCRYS*HID + 255)/256, 256, 0, stream>>>(
      pre, comb, meanh, rstdh, gfc, betafc, BCRYS*HID, 128);

  // VAE heads
  heads_kernel<<<BCRYS, 64, 0, stream>>>(comb, Wmu, bmu, Wlv, blv, eps, out_mu, out_lv, z);

  // latent -> features
  gemm64to128_kernel<<<BCRYS, 128, 0, stream>>>(z, Wl, bl, pre);
  hipMemsetAsync(part, 0, 64*256*4, stream);
  colstats_kernel<<<256, 256, 0, stream>>>(pre, BCRYS, 128, part);
  finalize_stats_kernel<<<2, 64, 0, stream>>>(part, meanh, rstdh, 128, (float)BCRYS);
  bn_relu_kernel<<<(BCRYS*HID + 255)/256, 256, 0, stream>>>(
      pre, sfeat, meanh, rstdh, gl, betal, BCRYS*HID, 128);

  // atom feature / type decoders
  af1_kernel<<<BCRYS*NA, 128, 0, stream>>>(sfeat, Wa1, ba1, af1);
  af2_kernel<<<BCRYS*NA, 64, 0, stream>>>(af1, Wa2, ba2, out_af, afh);
  t1_kernel<<<BCRYS*NA, 64, 0, stream>>>(out_af, Wt1, bt1, t1);
  at_softmax_kernel<<<BCRYS*NA, 128, 0, stream>>>(t1, Wt2, bt2, out_at);

  // bond decoder (fused WMMA x2)
  bonds_kernel<<<BCRYS*NPAIR/16, 128, 0, stream>>>(afh, pb1, bb1, pb2, bb2, out_bd);
}
